// KMeansQuantizer_58497454572171
// MI455X (gfx1250) — matrix operand order repaired and checked by reading, not compiled, not run on hardware
//
#include <hip/hip_runtime.h>

// ---------------------------------------------------------------------------
// VQ-VAE quantizer for MI455X (gfx1250, wave32).
// Distance GEMM (8192x8192x512) via v_wmma_f32_16x16x32_bf16.
// 64 rows/block: each B tile amortized over 4 row-tiles (4 WMMA per B load).
// ---------------------------------------------------------------------------

#define D_DIM 512
#define K_DIM 8192
#define N_ROWS 8192
#define ROWS_PER_BLOCK 64
#define ROW_TILES 4                    // 16-row WMMA tiles per wave
#define XS_STRIDE 520                  // 512 + 8 halfs pad: lane bank stride = 4
#define MOM 0.9f
#define OMM 0.1f
#define CCOST 0.25f
#define EPSF 1e-20f

typedef __attribute__((ext_vector_type(16))) __bf16 v16bf;
typedef __attribute__((ext_vector_type(8)))  __bf16 v8bf;
typedef __attribute__((ext_vector_type(8)))  float  v8f;

__device__ __forceinline__ unsigned short f32_to_bf16_rte(float f) {
    unsigned int u = __float_as_uint(f);
    unsigned int r = u + 0x7FFFu + ((u >> 16) & 1u);  // round-to-nearest-even
    return (unsigned short)(r >> 16);
}

// --- Kernel 1: per-row squared norm + bf16 conversion of x (1 wave per row) -
__global__ __launch_bounds__(256)
void vq_prep_x(const float* __restrict__ x, unsigned short* __restrict__ xb,
               float* __restrict__ rn) {
    const int wave = threadIdx.x >> 5;
    const int lane = threadIdx.x & 31;
    const int row  = blockIdx.x * 8 + wave;
    const float* xr = x + (size_t)row * D_DIM;
    unsigned short* xbr = xb + (size_t)row * D_DIM;
    float s = 0.f;
#pragma unroll
    for (int i = 0; i < D_DIM / 32; ++i) {
        int d = lane + 32 * i;
        float v = xr[d];
        s += v * v;
        xbr[d] = f32_to_bf16_rte(v);
    }
#pragma unroll
    for (int off = 16; off > 0; off >>= 1) s += __shfl_xor(s, off, 32);
    if (lane == 0) rn[row] = s;
}

// --- Kernel 2: per-column norm of e, transpose-convert e -> ebT (K,D) bf16,
//               init new_un = 0.9*un, zero counts / loss accumulator ---------
__global__ __launch_bounds__(256)
void vq_prep_e(const float* __restrict__ e, const float* __restrict__ un,
               unsigned short* __restrict__ ebT, float* __restrict__ cn,
               float* __restrict__ counts, float* __restrict__ lossacc,
               float* __restrict__ un_out) {
    const int k = blockIdx.x * blockDim.x + threadIdx.x;
    float s = 0.f;
    for (int d = 0; d < D_DIM; ++d) {
        float v = e[(size_t)d * K_DIM + k];          // coalesced across threads
        s += v * v;
        ebT[(size_t)k * D_DIM + d] = f32_to_bf16_rte(v);
        un_out[(size_t)d * K_DIM + k] = MOM * un[(size_t)d * K_DIM + k];
    }
    cn[k] = s;
    counts[k] = 0.f;
    if (k == 0) lossacc[0] = 0.f;
}

// --- Kernel 3: argmin over K via bf16 WMMA; 64 rows per block ----------------
// 8 waves each own 1/8 of the K tiles and all 4 row-tiles; each 32-byte B tile
// feeds 4 WMMAs (A re-read from padded LDS).
__global__ __launch_bounds__(256)
void vq_argmin_wmma(const unsigned short* __restrict__ xb,
                    const unsigned short* __restrict__ ebT,
                    const float* __restrict__ rn,
                    const float* __restrict__ cn,
                    int* __restrict__ idx_ws,
                    float* __restrict__ idx_out,
                    float* __restrict__ counts) {
    __shared__ unsigned short xs[ROWS_PER_BLOCK * XS_STRIDE];  // ~65 KB
    __shared__ float s_val[8][ROWS_PER_BLOCK];
    __shared__ int   s_idx[8][ROWS_PER_BLOCK];

    const int row0 = blockIdx.x * ROWS_PER_BLOCK;
    {   // cooperative LDS fill with padded row stride (520 halfs = 65 uint4)
        const uint4* src = (const uint4*)(xb + (size_t)row0 * D_DIM);
        uint4* dst = (uint4*)xs;
        for (int i = threadIdx.x; i < ROWS_PER_BLOCK * 64; i += 256) {
            int r = i >> 6, c = i & 63;
            dst[r * 65 + c] = src[r * 64 + c];
        }
    }
    __syncthreads();

    const int wave = threadIdx.x >> 5;
    const int lane = threadIdx.x & 31;
    const int nsub = lane & 15;       // B column within tile / A row index
    const int half = lane >> 4;       // lane half selects K-subrange

    float rnv[ROW_TILES][8];
#pragma unroll
    for (int rt = 0; rt < ROW_TILES; ++rt)
#pragma unroll
        for (int j = 0; j < 8; ++j)
            rnv[rt][j] = rn[row0 + rt * 16 + j + half * 8];

    float best[ROW_TILES][8]; int bidx[ROW_TILES][8];
#pragma unroll
    for (int rt = 0; rt < ROW_TILES; ++rt)
#pragma unroll
        for (int j = 0; j < 8; ++j) { best[rt][j] = 3.4e38f; bidx[rt][j] = 0; }

    for (int kt = wave; kt < K_DIM / 16; kt += 8) {
        const int k0 = kt * 16;
        const unsigned short* brow = ebT + (size_t)(k0 + nsub) * D_DIM;
        // prefetch this lane's B row for the next k-tile this wave will visit
        if (kt + 8 < K_DIM / 16)
            __builtin_prefetch(ebT + (size_t)(k0 + 128 + nsub) * D_DIM, 0, 0);

        v8f acc[ROW_TILES];
#pragma unroll
        for (int rt = 0; rt < ROW_TILES; ++rt) acc[rt] = (v8f){};

#pragma unroll 2
        for (int dc = 0; dc < D_DIM; dc += 32) {
            v16bf b = *(const v16bf*)(brow + dc + half * 16);
#pragma unroll
            for (int rt = 0; rt < ROW_TILES; ++rt) {
                const unsigned short* arow =
                    xs + (rt * 16 + nsub) * XS_STRIDE + dc + half * 16;
                v8bf a0 = *(const v8bf*)(arow);
                v8bf a1 = *(const v8bf*)(arow + 8);
                v16bf a;
#pragma unroll
                for (int i = 0; i < 8; ++i) { a[i] = a0[i]; a[8 + i] = a1[i]; }
                acc[rt] = __builtin_amdgcn_wmma_f32_16x16x32_bf16(
                    false, a, false, b, (short)0, acc[rt], false, false);
            }
        }
        const float cnv = cn[k0 + nsub];
#pragma unroll
        for (int rt = 0; rt < ROW_TILES; ++rt)
#pragma unroll
            for (int j = 0; j < 8; ++j) {
                float dist = rnv[rt][j] + cnv - 2.0f * acc[rt][j];
                if (dist < best[rt][j]) { best[rt][j] = dist; bidx[rt][j] = k0 + nsub; }
            }
    }

    // reduce across the 16 lanes sharing (half, j) -> per-row argmin
#pragma unroll
    for (int rt = 0; rt < ROW_TILES; ++rt)
#pragma unroll
        for (int j = 0; j < 8; ++j) {
#pragma unroll
            for (int off = 1; off < 16; off <<= 1) {
                float ov = __shfl_xor(best[rt][j], off, 32);
                int   oi = __shfl_xor(bidx[rt][j], off, 32);
                if (ov < best[rt][j] || (ov == best[rt][j] && oi < bidx[rt][j])) {
                    best[rt][j] = ov; bidx[rt][j] = oi;
                }
            }
        }
    if (nsub == 0) {
#pragma unroll
        for (int rt = 0; rt < ROW_TILES; ++rt)
#pragma unroll
            for (int j = 0; j < 8; ++j) {
                s_val[wave][rt * 16 + j + half * 8] = best[rt][j];
                s_idx[wave][rt * 16 + j + half * 8] = bidx[rt][j];
            }
    }
    __syncthreads();
    if (threadIdx.x < ROWS_PER_BLOCK) {
        const int r = threadIdx.x;
        float bv = s_val[0][r]; int bi = s_idx[0][r];
#pragma unroll
        for (int w = 1; w < 8; ++w) {
            float ov = s_val[w][r]; int oi = s_idx[w][r];
            if (ov < bv || (ov == bv && oi < bi)) { bv = ov; bi = oi; }
        }
        idx_ws[row0 + r]  = bi;
        idx_out[row0 + r] = (float)bi;
        atomicAdd(&counts[bi], 1.0f);
    }
}

// --- Kernel 4: gather quantized, straight-through out, loss, un scatter ------
__global__ __launch_bounds__(256)
void vq_quantize(const float* __restrict__ x, const float* __restrict__ e,
                 const int* __restrict__ idx_ws,
                 float* __restrict__ q_out, float* __restrict__ un_out,
                 float* __restrict__ lossacc) {
    __shared__ float red[256];
    const int i = blockIdx.x * 256 + threadIdx.x;
    const int n = i >> 9;              // D_DIM == 512
    const int d = i & (D_DIM - 1);
    const int k = idx_ws[n];
    const float xv = x[i];
    const float ev = e[(size_t)d * K_DIM + k];
    q_out[i] = xv + (ev - xv);                      // straight-through numerics
    atomicAdd(&un_out[(size_t)d * K_DIM + k], OMM * xv);
    const float diff = ev - xv;
    red[threadIdx.x] = diff * diff;
    __syncthreads();
    for (int s = 128; s > 0; s >>= 1) {
        if (threadIdx.x < s) red[threadIdx.x] += red[threadIdx.x + s];
        __syncthreads();
    }
    if (threadIdx.x == 0) atomicAdd(lossacc, red[0]);
}

// --- Kernel 5: EMA cluster sizes, perplexity, loss, stable_cs ----------------
__global__ __launch_bounds__(256)
void vq_finalize(const float* __restrict__ cs, const float* __restrict__ counts,
                 const float* __restrict__ lossacc,
                 float* __restrict__ newcs_out, float* __restrict__ stable_ws,
                 float* __restrict__ loss_out, float* __restrict__ perp_out) {
    __shared__ float sn[256], sh[256];
    __shared__ float ntot;
    float ln = 0.f, lh = 0.f;
    for (int k = threadIdx.x; k < K_DIM; k += 256) {
        float c   = counts[k];
        float ncs = OMM * c + MOM * cs[k];
        newcs_out[k] = ncs;
        ln += ncs;
        float p = c * (1.0f / (float)N_ROWS);
        lh += p * __logf(p + EPSF);
    }
    sn[threadIdx.x] = ln; sh[threadIdx.x] = lh;
    __syncthreads();
    for (int s = 128; s > 0; s >>= 1) {
        if (threadIdx.x < s) {
            sn[threadIdx.x] += sn[threadIdx.x + s];
            sh[threadIdx.x] += sh[threadIdx.x + s];
        }
        __syncthreads();
    }
    if (threadIdx.x == 0) {
        ntot = sn[0];
        loss_out[0] = CCOST * lossacc[0] * (1.0f / (float)(N_ROWS * D_DIM));
        perp_out[0] = __expf(-sh[0]);
    }
    __syncthreads();
    const float nt  = ntot;
    const float mul = nt / (nt + (float)K_DIM * EPSF);
    for (int k = threadIdx.x; k < K_DIM; k += 256)
        stable_ws[k] = (newcs_out[k] + EPSF) * mul;
}

// --- Kernel 6: new_e = new_un / stable_cs ------------------------------------
__global__ __launch_bounds__(256)
void vq_newe(const float* __restrict__ un_out, const float* __restrict__ stable_ws,
             float* __restrict__ newe_out) {
    const int i = blockIdx.x * 256 + threadIdx.x;
    const int k = i & (K_DIM - 1);
    newe_out[i] = un_out[i] / stable_ws[k];
}

// ---------------------------------------------------------------------------
extern "C" void kernel_launch(void* const* d_in, const int* in_sizes, int n_in,
                              void* d_out, int out_size, void* d_ws, size_t ws_size,
                              hipStream_t stream) {
    const float* x  = (const float*)d_in[0];   // (16,512,512)
    const float* e  = (const float*)d_in[1];   // (512,8192)
    const float* cs = (const float*)d_in[2];   // (8192,)
    const float* un = (const float*)d_in[3];   // (512,8192)
    float* out = (float*)d_out;

    // output layout (flat concat, reference return order)
    float* q_out     = out;                        // 4,194,304
    float* loss_out  = out + 4194304;              // 1
    float* perp_out  = out + 4194305;              // 1
    float* idx_out   = out + 4194306;              // 8,192 (as float)
    float* newe_out  = out + 4202498;              // 4,194,304
    float* newcs_out = out + 8396802;              // 8,192
    float* un_out    = out + 8404994;              // 4,194,304

    // workspace layout (~17 MB)
    char* ws = (char*)d_ws;
    unsigned short* xb  = (unsigned short*)(ws);                     // 8 MB
    unsigned short* ebT = (unsigned short*)(ws + 8u * 1024 * 1024);  // 8 MB
    float* rn      = (float*)(ws + 16u * 1024 * 1024);               // 32 KB
    float* cn      = rn + N_ROWS;
    float* counts  = cn + K_DIM;
    int*   idx_ws  = (int*)(counts + K_DIM);
    float* stable  = (float*)(idx_ws + N_ROWS);
    float* lossacc = stable + K_DIM;

    vq_prep_x<<<N_ROWS / 8, 256, 0, stream>>>(x, xb, rn);
    vq_prep_e<<<K_DIM / 256, 256, 0, stream>>>(e, un, ebT, cn, counts, lossacc, un_out);
    vq_argmin_wmma<<<N_ROWS / ROWS_PER_BLOCK, 256, 0, stream>>>(
        xb, ebT, rn, cn, idx_ws, idx_out, counts);
    vq_quantize<<<(N_ROWS * D_DIM) / 256, 256, 0, stream>>>(x, e, idx_ws, q_out, un_out, lossacc);
    vq_finalize<<<1, 256, 0, stream>>>(cs, counts, lossacc, newcs_out, stable, loss_out, perp_out);
    vq_newe<<<(D_DIM * K_DIM) / 256, 256, 0, stream>>>(un_out, stable, newe_out);
}
